// TaskOnlyModel_55645596287314
// MI455X (gfx1250) — compile-verified, matrix operand order checked by reading, not run on hardware
//
#include <hip/hip_runtime.h>
#include <hip/hip_bf16.h>
#include <math.h>

// ---------------------------------------------------------------------------
// CDNA5 (gfx1250) wave32 WMMA types
// ---------------------------------------------------------------------------
typedef __bf16 bf16;
typedef __attribute__((ext_vector_type(16))) __bf16 v16bf;
typedef __attribute__((ext_vector_type(8)))  __bf16 v8bf;
typedef __attribute__((ext_vector_type(8)))  float  v8f;

union Frag { v16bf v; v8bf h[2]; };

__device__ __forceinline__ v8f wmma_bf16(v16bf a, v16bf b, v8f c) {
  // D = A(16x32 bf16) * B(32x16 bf16) + C(16x16 f32)
  return __builtin_amdgcn_wmma_f32_16x16x32_bf16(
      /*neg_a=*/false, a, /*neg_b=*/false, b,
      /*c_mod=*/(short)0, c, /*reuse_a=*/false, /*reuse_b=*/false);
}

// Load a 16x32 bf16 A-fragment (or B^T-fragment) from a row-major [rows, K]
// bf16 matrix. Per ISA layout: lane (hi,l) holds row=l, K chunks
// [k0+8*hi .. +8) in elems 0..7 and [k0+16+8*hi .. +8) in elems 8..15.
__device__ __forceinline__ v16bf load_frag(const bf16* __restrict__ base,
                                           long long stride, int row, int k0,
                                           int hi) {
  Frag f;
  const bf16* p = base + (long long)row * stride + k0 + (hi << 3);
  f.h[0] = *(const v8bf*)(p);
  f.h[1] = *(const v8bf*)(p + 16);
  return f.v;
}

__device__ __forceinline__ float gelu_tanh(float x) {
  const float c = 0.7978845608028654f;
  float x3 = x * x * x;
  return 0.5f * x * (1.0f + tanhf(c * (x + 0.044715f * x3)));
}
__device__ __forceinline__ float leaky01(float x) {
  return x > 0.0f ? x : 0.01f * x;
}

// ---------------------------------------------------------------------------
// Generic bf16 WMMA GEMM: C[M,N] = act(A[M,K] @ Bt[N,K]^T + bias)
// Block = 256 threads = 8 waves laid out 4(M) x 2(N); each wave owns a
// 32x64 tile = 2x4 WMMA tiles (8 f32 accumulators). Per k-step a wave loads
// 2 A-frags + 4 B-frags (12 x b128) for 8 WMMAs, and the next k-step's
// fragments are preloaded before the current WMMA group so the global loads
// overlap the matrix pipe. Requires K % 32 == 0. Bt is pre-transposed [N,K].
// ---------------------------------------------------------------------------
__launch_bounds__(256)
__global__ void gemm_bf16_wmma(const bf16* __restrict__ A,
                               const bf16* __restrict__ Bt,
                               const float* __restrict__ bias,
                               float* __restrict__ Cf, bf16* __restrict__ Cb,
                               int M, int N, int K, int act) {
  const int wave = threadIdx.x >> 5;
  const int lane = threadIdx.x & 31;
  const int hi = lane >> 4;
  const int l = lane & 15;
  const int wm = wave >> 1;  // 0..3
  const int wn = wave & 1;   // 0..1
  const int m0 = blockIdx.y * 128 + wm * 32;
  const int n0 = blockIdx.x * 128 + wn * 64;

  int arow[2], brow[4];
#pragma unroll
  for (int i = 0; i < 2; ++i) {
    int r = m0 + i * 16 + l;
    arow[i] = r < M ? r : M - 1;  // clamp; OOB rows masked at store
  }
#pragma unroll
  for (int j = 0; j < 4; ++j) {
    int r = n0 + j * 16 + l;
    brow[j] = r < N ? r : N - 1;
  }

  v8f acc[2][4];
#pragma unroll
  for (int i = 0; i < 2; ++i)
#pragma unroll
    for (int j = 0; j < 4; ++j)
#pragma unroll
      for (int r = 0; r < 8; ++r) acc[i][j][r] = 0.0f;

  // prologue: fragments for k0 = 0
  v16bf af[2], bf_[4];
#pragma unroll
  for (int i = 0; i < 2; ++i) af[i] = load_frag(A, K, arow[i], 0, hi);
#pragma unroll
  for (int j = 0; j < 4; ++j) bf_[j] = load_frag(Bt, K, brow[j], 0, hi);

  for (int k0 = 0; k0 < K; k0 += 32) {
    // preload next k-step (re-load current on the final iteration; branch-free)
    int kn = k0 + 32 < K ? k0 + 32 : k0;
    v16bf naf[2], nbf[4];
#pragma unroll
    for (int i = 0; i < 2; ++i) naf[i] = load_frag(A, K, arow[i], kn, hi);
#pragma unroll
    for (int j = 0; j < 4; ++j) nbf[j] = load_frag(Bt, K, brow[j], kn, hi);

#pragma unroll
    for (int i = 0; i < 2; ++i)
#pragma unroll
      for (int j = 0; j < 4; ++j)
        acc[i][j] = wmma_bf16(af[i], bf_[j], acc[i][j]);

#pragma unroll
    for (int i = 0; i < 2; ++i) af[i] = naf[i];
#pragma unroll
    for (int j = 0; j < 4; ++j) bf_[j] = nbf[j];
  }

#pragma unroll
  for (int j = 0; j < 4; ++j) {
    int col = n0 + j * 16 + l;
    if (col >= N) continue;
    float bv = bias ? bias[col] : 0.0f;
#pragma unroll
    for (int i = 0; i < 2; ++i) {
#pragma unroll
      for (int r = 0; r < 8; ++r) {
        int row = m0 + i * 16 + r + 8 * hi;
        if (row >= M) continue;
        float v = acc[i][j][r] + bv;
        if (act == 1) v = gelu_tanh(v);
        else if (act == 2) v = leaky01(v);
        long long idx = (long long)row * N + col;
        if (Cf) Cf[idx] = v;
        if (Cb) Cb[idx] = (bf16)v;
      }
    }
  }
}

// ---------------------------------------------------------------------------
// Fused single-head attention for one (batch, head): S=128, DH=64.
// Scores + softmax + context entirely on-chip with WMMA, wave32 shuffles.
// ---------------------------------------------------------------------------
#define AS 128
#define ADH 64
#define AD 768
#define AH 12

__launch_bounds__(256)
__global__ void attention_wmma(const bf16* __restrict__ qb,
                               const bf16* __restrict__ kb,
                               const bf16* __restrict__ vb,
                               const float* __restrict__ emask,
                               bf16* __restrict__ ctxb) {
  const int bh = blockIdx.x;
  const int b = bh / AH;
  const int h = bh % AH;
  const int hc = h * ADH;

  __shared__ bf16 Vt[ADH][AS + 8];       // V^T: [d][j]
  __shared__ bf16 P[8][16][AS + 8];      // per-wave probs, row-major [i][j]

  for (int idx = threadIdx.x; idx < AS * ADH; idx += 256) {
    int j = idx >> 6;          // token
    int d = idx & 63;          // head dim (coalesced global read)
    Vt[d][j] = vb[((long long)(b * AS + j)) * AD + hc + d];
  }
  __syncthreads();

  const int wave = threadIdx.x >> 5;
  const int lane = threadIdx.x & 31;
  const int hi = lane >> 4;
  const int l = lane & 15;

  // Q fragments for this wave's 16 query rows (K = DH = 64 -> 2 chunks)
  const bf16* qrow = qb + ((long long)(b * AS + wave * 16 + l)) * AD + hc;
  Frag qf0, qf1;
  qf0.h[0] = *(const v8bf*)(qrow + 8 * hi);
  qf0.h[1] = *(const v8bf*)(qrow + 16 + 8 * hi);
  qf1.h[0] = *(const v8bf*)(qrow + 32 + 8 * hi);
  qf1.h[1] = *(const v8bf*)(qrow + 48 + 8 * hi);

  v8f sc[8];
#pragma unroll
  for (int t = 0; t < 8; ++t)
#pragma unroll
    for (int r = 0; r < 8; ++r) sc[t][r] = 0.0f;

#pragma unroll
  for (int t = 0; t < 8; ++t) {
    const bf16* krow = kb + ((long long)(b * AS + t * 16 + l)) * AD + hc;
    Frag kf;
    kf.h[0] = *(const v8bf*)(krow + 8 * hi);
    kf.h[1] = *(const v8bf*)(krow + 16 + 8 * hi);
    sc[t] = wmma_bf16(qf0.v, kf.v, sc[t]);
    kf.h[0] = *(const v8bf*)(krow + 32 + 8 * hi);
    kf.h[1] = *(const v8bf*)(krow + 48 + 8 * hi);
    sc[t] = wmma_bf16(qf1.v, kf.v, sc[t]);
  }

  // mask bias depends only on key column (lane N dimension)
  float biasv[8];
#pragma unroll
  for (int t = 0; t < 8; ++t)
    biasv[t] = (1.0f - emask[b * AS + t * 16 + l]) * -1e9f;

  const float scale = 0.125f;  // 1/sqrt(64)
#pragma unroll
  for (int r = 0; r < 8; ++r) {
    float m = -3.0e38f;
#pragma unroll
    for (int t = 0; t < 8; ++t) {
      float v = sc[t][r] * scale + biasv[t];
      sc[t][r] = v;
      m = fmaxf(m, v);
    }
    // reduce across the 16-lane N group (xor 1,2,4,8 preserves hi bit)
    m = fmaxf(m, __shfl_xor(m, 1, 32));
    m = fmaxf(m, __shfl_xor(m, 2, 32));
    m = fmaxf(m, __shfl_xor(m, 4, 32));
    m = fmaxf(m, __shfl_xor(m, 8, 32));
    float ssum = 0.0f;
#pragma unroll
    for (int t = 0; t < 8; ++t) {
      float p = __expf(sc[t][r] - m);
      sc[t][r] = p;
      ssum += p;
    }
    ssum += __shfl_xor(ssum, 1, 32);
    ssum += __shfl_xor(ssum, 2, 32);
    ssum += __shfl_xor(ssum, 4, 32);
    ssum += __shfl_xor(ssum, 8, 32);
    float rinv = 1.0f / ssum;
    int prow = r + 8 * hi;
#pragma unroll
    for (int t = 0; t < 8; ++t)
      P[wave][prow][t * 16 + l] = (bf16)(sc[t][r] * rinv);
  }
  __syncthreads();  // also covers the wave-local LDS RAW

  // ctx = P[16x128] @ V[128x64]; B^T rows come from Vt
  v8f cf[4];
#pragma unroll
  for (int n = 0; n < 4; ++n)
#pragma unroll
    for (int r = 0; r < 8; ++r) cf[n][r] = 0.0f;

#pragma unroll
  for (int kk = 0; kk < 4; ++kk) {
    int k0 = kk * 32;
    Frag af;
    af.h[0] = *(const v8bf*)(&P[wave][l][k0 + 8 * hi]);
    af.h[1] = *(const v8bf*)(&P[wave][l][k0 + 16 + 8 * hi]);
#pragma unroll
    for (int n = 0; n < 4; ++n) {
      Frag bfr;
      bfr.h[0] = *(const v8bf*)(&Vt[n * 16 + l][k0 + 8 * hi]);
      bfr.h[1] = *(const v8bf*)(&Vt[n * 16 + l][k0 + 16 + 8 * hi]);
      cf[n] = wmma_bf16(af.v, bfr.v, cf[n]);
    }
  }

#pragma unroll
  for (int n = 0; n < 4; ++n)
#pragma unroll
    for (int r = 0; r < 8; ++r)
      ctxb[((long long)(b * AS + wave * 16 + r + 8 * hi)) * AD + hc + n * 16 + l] =
          (bf16)(cf[n][r]);
}

// ---------------------------------------------------------------------------
// Elementwise / utility kernels
// ---------------------------------------------------------------------------
__global__ void embed_kernel(const int* __restrict__ ids,
                             const float* __restrict__ tok,
                             const float* __restrict__ pos,
                             float* __restrict__ h0, bf16* __restrict__ hb,
                             long long total, int D, int S) {
  long long i = (long long)blockIdx.x * blockDim.x + threadIdx.x;
  if (i >= total) return;
  long long t = i / D;
  int d = (int)(i % D);
  int s = (int)(t % S);
  float v = tok[(long long)ids[t] * D + d] + pos[(long long)s * D + d];
  h0[i] = v;
  hb[i] = (bf16)v;
}

__global__ void convert_bf16_kernel(const float* __restrict__ in,
                                    bf16* __restrict__ out, long long n) {
  long long i = (long long)blockIdx.x * blockDim.x + threadIdx.x;
  if (i < n) out[i] = (bf16)in[i];
}

__global__ void leaky_convert_kernel(const float* __restrict__ in,
                                     bf16* __restrict__ out, long long n) {
  long long i = (long long)blockIdx.x * blockDim.x + threadIdx.x;
  if (i < n) out[i] = (bf16)leaky01(in[i]);
}

// out[N,K] bf16 = transpose(in[K,N] f32)
__global__ void convert_transpose_kernel(const float* __restrict__ in,
                                         bf16* __restrict__ out, int K, int N) {
  long long i = (long long)blockIdx.x * blockDim.x + threadIdx.x;
  long long total = (long long)K * N;
  if (i >= total) return;
  int n = (int)(i / K);
  int k = (int)(i % K);
  out[i] = (bf16)in[(long long)k * N + n];
}

// Residual-add + LayerNorm over D=768 per row; 256 threads/row.
__launch_bounds__(256)
__global__ void add_ln_kernel(const float* __restrict__ x,
                              const float* __restrict__ y,
                              const float* __restrict__ g,
                              const float* __restrict__ bta,
                              float* __restrict__ outf, bf16* __restrict__ outb) {
  const int D = 768;
  __shared__ float red[256];
  long long row = blockIdx.x;
  const float* xr = x + row * D;
  const float* yr = y + row * D;
  float vals[3];
  float s = 0.0f;
#pragma unroll
  for (int j = 0; j < 3; ++j) {
    float v = xr[threadIdx.x + 256 * j] + yr[threadIdx.x + 256 * j];
    vals[j] = v;
    s += v;
  }
  red[threadIdx.x] = s;
  __syncthreads();
  for (int o = 128; o > 0; o >>= 1) {
    if (threadIdx.x < o) red[threadIdx.x] += red[threadIdx.x + o];
    __syncthreads();
  }
  float mean = red[0] * (1.0f / D);
  __syncthreads();
  float vs = 0.0f;
#pragma unroll
  for (int j = 0; j < 3; ++j) {
    float d = vals[j] - mean;
    vs += d * d;
  }
  red[threadIdx.x] = vs;
  __syncthreads();
  for (int o = 128; o > 0; o >>= 1) {
    if (threadIdx.x < o) red[threadIdx.x] += red[threadIdx.x + o];
    __syncthreads();
  }
  float rstd = rsqrtf(red[0] * (1.0f / D) + 1e-12f);
#pragma unroll
  for (int j = 0; j < 3; ++j) {
    int c = threadIdx.x + 256 * j;
    float v = (vals[j] - mean) * rstd * g[c] + bta[c];
    outf[row * D + c] = v;
    if (outb) outb[row * D + c] = (bf16)v;
  }
}

__global__ void meanpool_kernel(const float* __restrict__ h2,
                                bf16* __restrict__ outb, int B, int S, int D) {
  long long i = (long long)blockIdx.x * blockDim.x + threadIdx.x;
  if (i >= (long long)B * D) return;
  int b = (int)(i / D);
  int d = (int)(i % D);
  float s = 0.0f;
  for (int t = 0; t < S; ++t) s += h2[((long long)b * S + t) * D + d];
  outb[i] = (bf16)(s * (1.0f / S));
}

// ---------------------------------------------------------------------------
// GCN helpers
// ---------------------------------------------------------------------------
__global__ void deg_init_kernel(float* deg, int n) {
  int i = blockIdx.x * blockDim.x + threadIdx.x;
  if (i < n) deg[i] = 1.0f;  // self-loop
}
__global__ void deg_count_kernel(const int* __restrict__ dst, float* deg,
                                 long long e) {
  long long i = (long long)blockIdx.x * blockDim.x + threadIdx.x;
  if (i < e) atomicAdd(&deg[dst[i]], 1.0f);
}
__global__ void deg_rsqrt_kernel(const float* deg, float* dinv, int n) {
  int i = blockIdx.x * blockDim.x + threadIdx.x;
  if (i < n) dinv[i] = rsqrtf(deg[i]);
}
// init out with self-loop term + bias
__global__ void gcn_self_kernel(const float* __restrict__ xw,
                                const float* __restrict__ dinv,
                                const float* __restrict__ bias,
                                float* __restrict__ out, long long total, int C) {
  long long i = (long long)blockIdx.x * blockDim.x + threadIdx.x;
  if (i >= total) return;
  int node = (int)(i / C);
  int c = (int)(i % C);
  float di = dinv[node];
  out[i] = xw[i] * di * di + bias[c];
}
__global__ void gcn_edge_kernel(const int* __restrict__ src,
                                const int* __restrict__ dst,
                                const float* __restrict__ xw,
                                const float* __restrict__ dinv,
                                float* __restrict__ out, long long total, int C) {
  long long i = (long long)blockIdx.x * blockDim.x + threadIdx.x;
  if (i >= total) return;
  long long e = i / C;
  int c = (int)(i % C);
  int s = src[e];
  int d = dst[e];
  atomicAdd(&out[(long long)d * C + c],
            xw[(long long)s * C + c] * dinv[s] * dinv[d]);
}

__global__ void gather_concat_kernel(const float* __restrict__ g2,
                                     const int* __restrict__ bidx,
                                     const bf16* __restrict__ xtextb,
                                     float* __restrict__ xg_out,
                                     bf16* __restrict__ xcb, int B, int CS) {
  int i = blockIdx.x * blockDim.x + threadIdx.x;
  if (i >= B * CS) return;
  int b = i / CS;
  int c = i % CS;
  float v = g2[(long long)bidx[b] * CS + c];
  xg_out[i] = v;
  xcb[(long long)b * 2 * CS + c] = xtextb[i];
  xcb[(long long)b * 2 * CS + CS + c] = (bf16)v;
}

// ---------------------------------------------------------------------------
// Host orchestration
// ---------------------------------------------------------------------------
extern "C" void kernel_launch(void* const* d_in, const int* in_sizes, int n_in,
                              void* d_out, int out_size, void* d_ws,
                              size_t ws_size, hipStream_t stream) {
  const int B = 512, S = 128, D = 768, FF = 3072, CS = 128, NC = 16;
  const int NN = 50000;
  const long long E = 800000;
  const long long T = (long long)B * S;

  const int* e_ids = (const int*)d_in[0];
  const float* e_mask = (const float*)d_in[1];
  const float* x_graph = (const float*)d_in[2];
  const int* edge_index = (const int*)d_in[3];  // [2, E]
  const int* batch_idx = (const int*)d_in[4];
  const float* tok_emb = (const float*)d_in[6];
  const float* pos_emb = (const float*)d_in[7];
  const float* wq = (const float*)d_in[8];  const float* bq = (const float*)d_in[9];
  const float* wk = (const float*)d_in[10]; const float* bk = (const float*)d_in[11];
  const float* wv = (const float*)d_in[12]; const float* bv = (const float*)d_in[13];
  const float* wo = (const float*)d_in[14]; const float* bo = (const float*)d_in[15];
  const float* ln1_g = (const float*)d_in[16]; const float* ln1_b = (const float*)d_in[17];
  const float* w_ff1 = (const float*)d_in[18]; const float* b_ff1 = (const float*)d_in[19];
  const float* w_ff2 = (const float*)d_in[20]; const float* b_ff2 = (const float*)d_in[21];
  const float* ln2_g = (const float*)d_in[22]; const float* ln2_b = (const float*)d_in[23];
  const float* w_text = (const float*)d_in[24]; const float* b_text = (const float*)d_in[25];
  const float* w_g1 = (const float*)d_in[26]; const float* b_g1 = (const float*)d_in[27];
  const float* w_g2 = (const float*)d_in[28]; const float* b_g2 = (const float*)d_in[29];
  const float* w_c1 = (const float*)d_in[30]; const float* b_c1 = (const float*)d_in[31];
  const float* w_c2 = (const float*)d_in[32]; const float* b_c2 = (const float*)d_in[33];

  float* out_logits = (float*)d_out;
  float* out_xtext = out_logits + (size_t)B * NC;
  float* out_xg = out_xtext + (size_t)B * CS;

  // ---- workspace bump allocation (aliased across phases) ----
  char* ws = (char*)d_ws;
  size_t off = 0;
  auto alloc = [&](size_t bytes) -> char* {
    char* p = ws + off;
    off += (bytes + 255) & ~(size_t)255;
    return p;
  };
  float* h0 = (float*)alloc(T * D * 4);
  bf16* hb = (bf16*)alloc(T * D * 2);
  bf16* qb = (bf16*)alloc(T * D * 2);
  bf16* kb = (bf16*)alloc(T * D * 2);
  bf16* vb = (bf16*)alloc(T * D * 2);
  float* attnout = (float*)alloc(T * D * 4);
  float* h1 = (float*)alloc(T * D * 4);
  char* big = alloc((size_t)T * FF * 2);  // ff1b, later GCN arena

  bf16* ctxb = hb;     // hb dead after QKV GEMMs
  bf16* h1b = qb;      // qb dead after attention
  bf16* ff1b = (bf16*)big;
  float* ff2 = attnout;  // attnout dead after ln1
  float* h2 = h0;        // h0 dead after ln1

  // GCN arena inside `big` (ff1b dead after ff2 GEMM)
  size_t go = 0;
  auto galloc = [&](size_t bytes) -> char* {
    char* p = big + go;
    go += (bytes + 255) & ~(size_t)255;
    return p;
  };
  bf16* xgb = (bf16*)galloc((size_t)NN * D * 2);
  float* xw1 = (float*)galloc((size_t)NN * 2 * CS * 4);
  float* g1 = (float*)galloc((size_t)NN * 2 * CS * 4);
  bf16* g1b = (bf16*)galloc((size_t)NN * 2 * CS * 2);
  float* xw2 = (float*)galloc((size_t)NN * CS * 4);
  float* g2 = (float*)galloc((size_t)NN * CS * 4);

  // small always-live region
  bf16* wqT = (bf16*)alloc((size_t)D * D * 2);
  bf16* wkT = (bf16*)alloc((size_t)D * D * 2);
  bf16* wvT = (bf16*)alloc((size_t)D * D * 2);
  bf16* woT = (bf16*)alloc((size_t)D * D * 2);
  bf16* wff1T = (bf16*)alloc((size_t)D * FF * 2);
  bf16* wff2T = (bf16*)alloc((size_t)D * FF * 2);
  bf16* wtextT = (bf16*)alloc((size_t)D * CS * 2);
  bf16* wg1T = (bf16*)alloc((size_t)D * 2 * CS * 2);
  bf16* wg2T = (bf16*)alloc((size_t)2 * CS * CS * 2);
  bf16* wc1T = (bf16*)alloc((size_t)2 * CS * CS * 2);
  bf16* wc2T = (bf16*)alloc((size_t)CS * NC * 2);
  float* deg = (float*)alloc((size_t)NN * 4);
  float* dinv = (float*)alloc((size_t)NN * 4);
  bf16* xtpre = (bf16*)alloc((size_t)B * D * 2);
  bf16* xtextb = (bf16*)alloc((size_t)B * CS * 2);
  bf16* xcb = (bf16*)alloc((size_t)B * 2 * CS * 2);
  bf16* hcb = (bf16*)alloc((size_t)B * CS * 2);
  (void)n_in; (void)in_sizes; (void)out_size; (void)ws_size;

  auto blocks = [](long long total) -> unsigned {
    return (unsigned)((total + 255) / 256);
  };
  auto convT = [&](const float* w, bf16* wT, int K, int N) {
    convert_transpose_kernel<<<blocks((long long)K * N), 256, 0, stream>>>(w, wT, K, N);
  };
  auto gemm = [&](const bf16* A, const bf16* Bt, const float* bias, float* Cf,
                  bf16* Cb, int M, int N, int K, int act) {
    dim3 g((N + 127) / 128, (M + 127) / 128);
    gemm_bf16_wmma<<<g, 256, 0, stream>>>(A, Bt, bias, Cf, Cb, M, N, K, act);
  };

  // ---- weight conversion / transposition (bf16, [N,K] row-major) ----
  convT(wq, wqT, D, D);       convT(wk, wkT, D, D);
  convT(wv, wvT, D, D);       convT(wo, woT, D, D);
  convT(w_ff1, wff1T, D, FF); convT(w_ff2, wff2T, FF, D);
  convT(w_text, wtextT, D, CS);
  convT(w_g1, wg1T, D, 2 * CS);
  convT(w_g2, wg2T, 2 * CS, CS);
  convT(w_c1, wc1T, 2 * CS, CS);
  convT(w_c2, wc2T, CS, NC);

  // ---- transformer ----
  embed_kernel<<<blocks(T * D), 256, 0, stream>>>(e_ids, tok_emb, pos_emb, h0,
                                                  hb, T * D, D, S);
  gemm(hb, wqT, bq, nullptr, qb, (int)T, D, D, 0);
  gemm(hb, wkT, bk, nullptr, kb, (int)T, D, D, 0);
  gemm(hb, wvT, bv, nullptr, vb, (int)T, D, D, 0);
  attention_wmma<<<B * AH, 256, 0, stream>>>(qb, kb, vb, e_mask, ctxb);
  gemm(ctxb, woT, bo, attnout, nullptr, (int)T, D, D, 0);
  add_ln_kernel<<<(unsigned)T, 256, 0, stream>>>(h0, attnout, ln1_g, ln1_b, h1, h1b);
  gemm(h1b, wff1T, b_ff1, nullptr, ff1b, (int)T, FF, D, /*gelu*/ 1);
  gemm(ff1b, wff2T, b_ff2, ff2, nullptr, (int)T, D, FF, 0);
  add_ln_kernel<<<(unsigned)T, 256, 0, stream>>>(h1, ff2, ln2_g, ln2_b, h2, nullptr);

  // ---- mean pool + text MLP (x_text written straight to d_out) ----
  meanpool_kernel<<<blocks((long long)B * D), 256, 0, stream>>>(h2, xtpre, B, S, D);
  gemm(xtpre, wtextT, b_text, out_xtext, xtextb, B, CS, D, /*leaky*/ 2);

  // ---- GCN layer 1 ----
  convert_bf16_kernel<<<blocks((long long)NN * D), 256, 0, stream>>>(x_graph, xgb, (long long)NN * D);
  gemm(xgb, wg1T, nullptr, xw1, nullptr, NN, 2 * CS, D, 0);
  deg_init_kernel<<<blocks(NN), 256, 0, stream>>>(deg, NN);
  deg_count_kernel<<<blocks(E), 256, 0, stream>>>(edge_index + E, deg, E);
  deg_rsqrt_kernel<<<blocks(NN), 256, 0, stream>>>(deg, dinv, NN);
  gcn_self_kernel<<<blocks((long long)NN * 2 * CS), 256, 0, stream>>>(
      xw1, dinv, b_g1, g1, (long long)NN * 2 * CS, 2 * CS);
  gcn_edge_kernel<<<blocks(E * 2 * CS), 256, 0, stream>>>(
      edge_index, edge_index + E, xw1, dinv, g1, E * 2 * CS, 2 * CS);
  leaky_convert_kernel<<<blocks((long long)NN * 2 * CS), 256, 0, stream>>>(
      g1, g1b, (long long)NN * 2 * CS);

  // ---- GCN layer 2 ----
  gemm(g1b, wg2T, nullptr, xw2, nullptr, NN, CS, 2 * CS, 0);
  gcn_self_kernel<<<blocks((long long)NN * CS), 256, 0, stream>>>(
      xw2, dinv, b_g2, g2, (long long)NN * CS, CS);
  gcn_edge_kernel<<<blocks(E * CS), 256, 0, stream>>>(
      edge_index, edge_index + E, xw2, dinv, g2, E * CS, CS);

  // ---- gather / concat / classifier ----
  gather_concat_kernel<<<blocks((long long)B * CS), 256, 0, stream>>>(
      g2, batch_idx, xtextb, out_xg, xcb, B, CS);
  gemm(xcb, wc1T, b_c1, nullptr, hcb, B, CS, 2 * CS, /*leaky*/ 2);
  gemm(hcb, wc2T, b_c2, out_logits, nullptr, B, NC, CS, 0);
}
